// RecSSM_47863115547036
// MI455X (gfx1250) — matrix operand order repaired and checked by reading, not compiled, not run on hardware
//
#include <hip/hip_runtime.h>
#include <math.h>

// ---------------- CDNA5 types / helpers ----------------
typedef __attribute__((ext_vector_type(16))) __bf16 v16bf;
typedef __attribute__((ext_vector_type(8)))  float  v8f;

__device__ __forceinline__ __bf16 tobf(float f) {
  unsigned u = __builtin_bit_cast(unsigned, f);
  u += 0x7fffu + ((u >> 16) & 1u);              // round-to-nearest-even
  unsigned short h = (unsigned short)(u >> 16);
  return __builtin_bit_cast(__bf16, h);
}
// fast SiLU: v_exp_f32 + v_rcp_f32 (no IEEE div expansion)
__device__ __forceinline__ float silu_f(float x) {
  return x * __builtin_amdgcn_rcpf(1.0f + __expf(-x));
}

// A-fragment element -> K (contiguous order): lanes 0-15 K0-15, lanes 16-31 K16-31
__device__ __forceinline__ int a_elem_k(int e, int kadd) {
  int j = e >> 1, p = e & 1;
  return (2 * j + p) + 2 * kadd;
}

// =======================================================
// Conv block 1: [512,3,128,128] -> conv3x3(6) + BN + SiLU + maxpool4 -> [512,6,32,32]
// One WG = one batch, one 4-row band (pre-pool). 8 waves, each wave: 16-col strip x 4 rows.
// Implicit GEMM: M=16 pixels, N=16 (6 used), K=32 (27 used), 1 WMMA per tile.
// =======================================================
__global__ __launch_bounds__(256)
void conv1_kernel(const float* __restrict__ x, const float* __restrict__ w,
                  const float* __restrict__ b, const float* __restrict__ g,
                  const float* __restrict__ be, const float* __restrict__ mn,
                  const float* __restrict__ vr, float* __restrict__ out) {
  __shared__ float s_in[3][6][132];      // rows band*4-1 .. band*4+4, cols -1..130
  __shared__ float s_act[4][128][6];     // pre-pool activations
  __shared__ float s_scale[6], s_shift[6];

  const int band = blockIdx.x;           // pooled output row 0..31
  const int n    = blockIdx.y;
  const int tid  = threadIdx.x;
  const int lane = tid & 31, wave = tid >> 5;

  // prefetch the band's input rows into L2/WGP$ (global_prefetch_b8)
  {
    int r = tid % 6, ci = (tid / 6) % 3;
    int iy = band * 4 - 1 + r;
    if (tid < 18 && iy >= 0 && iy < 128)
      __builtin_prefetch(&x[((n * 3 + ci) * 128 + iy) * 128], 0, 3);
  }
  if (tid < 6) {
    float sc = g[tid] * rsqrtf(vr[tid] + 1e-5f);
    s_scale[tid] = sc;
    s_shift[tid] = be[tid] + (b[tid] - mn[tid]) * sc;
  }
  for (int i = tid; i < 3 * 6 * 132; i += 256) {
    int ci = i / (6 * 132), r = (i / 132) % 6, c = i % 132;
    int iy = band * 4 - 1 + r, ix = c - 1;
    float val = 0.0f;
    if (iy >= 0 && iy < 128 && ix >= 0 && ix < 128)
      val = x[((n * 3 + ci) * 128 + iy) * 128 + ix];
    s_in[ci][r][c] = val;
  }
  __syncthreads();

  // B fragment (weights 27x6, padded 32x16): lane = column, lanes 0-15 K0-15, 16-31 K16-31
  v16bf bfrag;
  {
    int ncol = lane & 15;
    int kb = (lane < 16) ? 0 : 16;
#pragma unroll
    for (int e = 0; e < 16; ++e) {
      int K = kb + e;
      float wv = (ncol < 6 && K < 27) ? w[ncol * 27 + K] : 0.0f;
      bfrag[e] = tobf(wv);
    }
  }

  const int colbase = wave * 16;
  const int kadd = (lane < 16) ? 0 : 8;
  for (int rl = 0; rl < 4; ++rl) {
    v16bf afrag;
    int ox = colbase + (lane & 15);
#pragma unroll
    for (int e = 0; e < 16; ++e) {
      int K = a_elem_k(e, kadd);
      float av = 0.0f;
      if (K < 27) {
        int ci = K / 9, rem = K % 9, ky = rem / 3, kx = rem % 3;
        av = s_in[ci][rl + ky][ox + kx];
      }
      afrag[e] = tobf(av);
    }
    v8f acc = {};
    acc = __builtin_amdgcn_wmma_f32_16x16x32_bf16(false, afrag, false, bfrag,
                                                  (short)0, acc, false, false);
    int nch = lane & 15;
    int mbase = (lane < 16) ? 0 : 8;
    if (nch < 6) {
      float sc = s_scale[nch], sh = s_shift[nch];
#pragma unroll
      for (int r = 0; r < 8; ++r)
        s_act[rl][colbase + mbase + r][nch] = silu_f(acc[r] * sc + sh);
    }
  }
  __syncthreads();

  if (tid < 6 * 32) {                    // 4x4 maxpool -> one pooled row
    int ch = tid >> 5, pc = tid & 31;
    float mx = -1e30f;
    for (int r = 0; r < 4; ++r)
      for (int c = 0; c < 4; ++c)
        mx = fmaxf(mx, s_act[r][pc * 4 + c][ch]);
    out[((n * 6 + ch) * 32 + band) * 32 + pc] = mx;
  }
}

// =======================================================
// Conv block 2: [512,6,32,32] -> conv3x3(12)+BN+SiLU+pool4 -> [512,12,8,8]
// One WG = one image. K=54 -> 2 k-steps, N=12 padded to 16.
// =======================================================
__global__ __launch_bounds__(256)
void conv2_kernel(const float* __restrict__ x, const float* __restrict__ w,
                  const float* __restrict__ b, const float* __restrict__ g,
                  const float* __restrict__ be, const float* __restrict__ mn,
                  const float* __restrict__ vr, float* __restrict__ out) {
  __shared__ float s_in[6][34][34];
  __shared__ float s_act[32][32][12];
  __shared__ float s_scale[12], s_shift[12];

  const int n = blockIdx.x;
  const int tid = threadIdx.x;
  const int lane = tid & 31, wave = tid >> 5;

  if (tid < 6)
    __builtin_prefetch(&x[(n * 6 + tid) * 32 * 32], 0, 3);
  if (tid < 12) {
    float sc = g[tid] * rsqrtf(vr[tid] + 1e-5f);
    s_scale[tid] = sc;
    s_shift[tid] = be[tid] + (b[tid] - mn[tid]) * sc;
  }
  for (int i = tid; i < 6 * 34 * 34; i += 256) {
    int ci = i / (34 * 34), r = (i / 34) % 34, c = i % 34;
    int iy = r - 1, ix = c - 1;
    float val = 0.0f;
    if (iy >= 0 && iy < 32 && ix >= 0 && ix < 32)
      val = x[((n * 6 + ci) * 32 + iy) * 32 + ix];
    s_in[ci][r][c] = val;
  }
  __syncthreads();

  v16bf bfrag[2];
  {
    int ncol = lane & 15;
    int kb = (lane < 16) ? 0 : 16;
    for (int ks = 0; ks < 2; ++ks) {
#pragma unroll
      for (int e = 0; e < 16; ++e) {
        int K = ks * 32 + kb + e;
        float wv = (ncol < 12 && K < 54) ? w[ncol * 54 + K] : 0.0f;
        bfrag[ks][e] = tobf(wv);
      }
    }
  }

  const int kadd = (lane < 16) ? 0 : 8;
  for (int rl = 0; rl < 4; ++rl) {
    int row = wave * 4 + rl;
    for (int tb = 0; tb < 2; ++tb) {
      int colbase = tb * 16;
      int ox = colbase + (lane & 15);
      v8f acc = {};
      for (int ks = 0; ks < 2; ++ks) {
        v16bf afrag;
#pragma unroll
        for (int e = 0; e < 16; ++e) {
          int K = ks * 32 + a_elem_k(e, kadd);
          float av = 0.0f;
          if (K < 54) {
            int ci = K / 9, rem = K % 9, ky = rem / 3, kx = rem % 3;
            av = s_in[ci][row + ky][ox + kx];
          }
          afrag[e] = tobf(av);
        }
        acc = __builtin_amdgcn_wmma_f32_16x16x32_bf16(false, afrag, false, bfrag[ks],
                                                      (short)0, acc, false, false);
      }
      int nch = lane & 15;
      int mbase = (lane < 16) ? 0 : 8;
      if (nch < 12) {
        float sc = s_scale[nch], sh = s_shift[nch];
#pragma unroll
        for (int r = 0; r < 8; ++r)
          s_act[row][colbase + mbase + r][nch] = silu_f(acc[r] * sc + sh);
      }
    }
  }
  __syncthreads();

  for (int i = tid; i < 12 * 8 * 8; i += 256) {
    int ch = i / 64, py = (i / 8) % 8, px = i % 8;
    float mx = -1e30f;
    for (int r = 0; r < 4; ++r)
      for (int c = 0; c < 4; ++c)
        mx = fmaxf(mx, s_act[py * 4 + r][px * 4 + c][ch]);
    out[((n * 12 + ch) * 8 + py) * 8 + px] = mx;
  }
}

// =======================================================
// Conv block 3: [512,12,8,8] -> conv3x3(24)+BN+SiLU+pool4 -> [512,24,2,2] -> [512,24,4]
// One WG = one image. K=108 -> 4 k-steps; N=24 -> 2 N-tiles. 8 waves, one (M,N) tile each.
// =======================================================
__global__ __launch_bounds__(256)
void conv3_kernel(const float* __restrict__ x, const float* __restrict__ w,
                  const float* __restrict__ b, const float* __restrict__ g,
                  const float* __restrict__ be, const float* __restrict__ mn,
                  const float* __restrict__ vr, float* __restrict__ out) {
  __shared__ float s_in[12][10][10];
  __shared__ float s_act[24][64];        // [channel][pixel]
  __shared__ float s_scale[24], s_shift[24];

  const int n = blockIdx.x;
  const int tid = threadIdx.x;
  const int lane = tid & 31, wave = tid >> 5;

  if (tid < 12)
    __builtin_prefetch(&x[(n * 12 + tid) * 64], 0, 3);
  if (tid < 24) {
    float sc = g[tid] * rsqrtf(vr[tid] + 1e-5f);
    s_scale[tid] = sc;
    s_shift[tid] = be[tid] + (b[tid] - mn[tid]) * sc;
  }
  for (int i = tid; i < 12 * 10 * 10; i += 256) {
    int ci = i / 100, r = (i / 10) % 10, c = i % 10;
    int iy = r - 1, ix = c - 1;
    float val = 0.0f;
    if (iy >= 0 && iy < 8 && ix >= 0 && ix < 8)
      val = x[((n * 12 + ci) * 8 + iy) * 8 + ix];
    s_in[ci][r][c] = val;
  }
  __syncthreads();

  const int mt = wave & 3;               // M tile (16 pixels)
  const int nt = wave >> 2;              // N tile (16 channels)
  const int kadd = (lane < 16) ? 0 : 8;
  const int kb = (lane < 16) ? 0 : 16;
  const int pix = mt * 16 + (lane & 15);
  const int prow = pix >> 3, pcol = pix & 7;

  v8f acc = {};
  for (int ks = 0; ks < 4; ++ks) {
    v16bf afrag, bfrag;
#pragma unroll
    for (int e = 0; e < 16; ++e) {
      int K = ks * 32 + a_elem_k(e, kadd);
      float av = 0.0f;
      if (K < 108) {
        int ci = K / 9, rem = K % 9, ky = rem / 3, kx = rem % 3;
        av = s_in[ci][prow + ky][pcol + kx];
      }
      afrag[e] = tobf(av);
    }
    int ncol = nt * 16 + (lane & 15);
#pragma unroll
    for (int e = 0; e < 16; ++e) {
      int K = ks * 32 + kb + e;
      float wv = (ncol < 24 && K < 108) ? w[ncol * 108 + K] : 0.0f;
      bfrag[e] = tobf(wv);
    }
    acc = __builtin_amdgcn_wmma_f32_16x16x32_bf16(false, afrag, false, bfrag,
                                                  (short)0, acc, false, false);
  }
  {
    int nch = nt * 16 + (lane & 15);
    int mbase = (lane < 16) ? 0 : 8;
    if (nch < 24) {
      float sc = s_scale[nch], sh = s_shift[nch];
#pragma unroll
      for (int r = 0; r < 8; ++r)
        s_act[nch][mt * 16 + mbase + r] = silu_f(acc[r] * sc + sh);
    }
  }
  __syncthreads();

  if (tid < 24 * 4) {                    // pool 4x4 -> [24][2][2]; reshape to [24][4]
    int ch = tid >> 2, py = (tid >> 1) & 1, px = tid & 1;
    float mx = -1e30f;
    for (int r = 0; r < 4; ++r)
      for (int c = 0; c < 4; ++c)
        mx = fmaxf(mx, s_act[ch][(py * 4 + r) * 8 + px * 4 + c]);
    out[(n * 24 + ch) * 4 + py * 2 + px] = mx;
  }
}

// =======================================================
// SSM stack: 3x (mamba2 + conv1d + BN + SiLU) + repeat(4) + L2 norm.
// One wave per batch element; scan state striped over lanes by d_state (2 per lane);
// cross-lane reductions via wave32 shfl_xor; all intermediates in LDS.
// =======================================================
struct SsmParams {
  const float *W_in, *conv_w, *conv_b, *dt_bias, *A_log, *D, *norm_w, *W_out;
  const float *cw, *cb, *g, *be, *m, *v;
  int T, cout;
};
struct SsmAll { SsmParams blk[3]; };

__global__ __launch_bounds__(32)
void ssm_kernel(const float* __restrict__ y3, float* __restrict__ outp, SsmAll P) {
  __shared__ float s_x[24 * 4];
  __shared__ float s_z[24 * 148];
  __shared__ float s_xbc[24 * 136];
  __shared__ float s_y[24 * 8];
  __shared__ float s_x2[24 * 4];

  const int n = blockIdx.x, tid = threadIdx.x;
  for (int i = tid; i < 24 * 4; i += 32) s_x[i] = y3[n * 96 + i];
  __syncthreads();

  for (int blk = 0; blk < 3; ++blk) {
    const SsmParams& pp = P.blk[blk];
    const int T = pp.T;

    // in-projection: zxbcdt[t,c] = sum_d x[t,d] * W_in[d,c]
    for (int i = tid; i < T * 148; i += 32) {
      int t = i / 148, c = i % 148;
      float s = 0.0f;
      for (int d = 0; d < 4; ++d) s += s_x[t * 4 + d] * pp.W_in[d * 148 + c];
      s_z[i] = s;
    }
    __syncthreads();

    // causal depthwise conv (K=4) + SiLU on xBC (channels 8..143)
    for (int i = tid; i < T * 136; i += 32) {
      int t = i / 136, cc = i % 136;
      float s = pp.conv_b[cc];
      for (int k = 0; k < 4; ++k) {
        int ts = t + k - 3;
        if (ts >= 0) s += s_z[ts * 148 + 8 + cc] * pp.conv_w[cc * 4 + k];
      }
      s_xbc[i] = silu_f(s);
    }
    __syncthreads();

    // selective scan: lane holds state columns n0=2*tid, n1=2*tid+1 for all (h,p)
    float st[4][2][2] = {};
    float Ah[4];
    for (int h = 0; h < 4; ++h) Ah[h] = -__expf(pp.A_log[h]);
    const int n0 = 2 * tid, n1 = 2 * tid + 1;
    for (int t = 0; t < T; ++t) {
      float Bv0 = s_xbc[t * 136 + 8 + n0], Bv1 = s_xbc[t * 136 + 8 + n1];
      float Cv0 = s_xbc[t * 136 + 72 + n0], Cv1 = s_xbc[t * 136 + 72 + n1];
#pragma unroll
      for (int h = 0; h < 4; ++h) {
        float dtr = s_z[t * 148 + 144 + h] + pp.dt_bias[h];
        float dt = (dtr > 20.0f) ? dtr : log1pf(__expf(dtr));
        float dA = __expf(dt * Ah[h]);
#pragma unroll
        for (int p = 0; p < 2; ++p) {
          float xv = s_xbc[t * 136 + h * 2 + p];
          float dBx = dt * xv;
          st[h][p][0] = st[h][p][0] * dA + dBx * Bv0;
          st[h][p][1] = st[h][p][1] * dA + dBx * Bv1;
          float part = st[h][p][0] * Cv0 + st[h][p][1] * Cv1;
          for (int off = 16; off > 0; off >>= 1) part += __shfl_xor(part, off, 32);
          if (tid == 0) s_y[t * 8 + h * 2 + p] = part + pp.D[h] * xv;
        }
      }
    }
    __syncthreads();

    // gate with silu(z), RMSNormGated, out-projection
    if (tid < T) {
      float yv[8], ss = 0.0f;
      for (int i = 0; i < 8; ++i) {
        float val = s_y[tid * 8 + i] * silu_f(s_z[tid * 148 + i]);
        yv[i] = val; ss += val * val;
      }
      float r = rsqrtf(ss / 8.0f + 1e-5f);
      for (int i = 0; i < 8; ++i) yv[i] *= r * pp.norm_w[i];
      for (int d = 0; d < 4; ++d) {
        float s = 0.0f;
        for (int i = 0; i < 8; ++i) s += yv[i] * pp.W_out[i * 4 + d];
        s_x2[tid * 4 + d] = s;
      }
    }
    __syncthreads();

    // conv1d over spatial (4) with cin=T, + BN + SiLU -> next block input [cout][4]
    const int cout = pp.cout;
    for (int i = tid; i < cout * 4; i += 32) {
      int co = i / 4, h = i % 4;
      float s = pp.cb[co];
      for (int ci = 0; ci < T; ++ci)
        for (int k = 0; k < 3; ++k) {
          int hh = h + k - 1;
          if (hh >= 0 && hh < 4) s += s_x2[ci * 4 + hh] * pp.cw[(co * T + ci) * 3 + k];
        }
      float sc = pp.g[co] * rsqrtf(pp.v[co] + 1e-5f);
      s_x[i] = silu_f((s - pp.m[co]) * sc + pp.be[co]);
    }
    __syncthreads();
  }

  // repeat(4) along last dim -> [8][16] -> flat [128], then L2 normalize
  float vals[4], ss = 0.0f;
#pragma unroll
  for (int j = 0; j < 4; ++j) {
    int idx = tid * 4 + j;
    int c = idx >> 4, q = idx & 15;
    float vv = s_x[c * 4 + (q >> 2)];
    vals[j] = vv; ss += vv * vv;
  }
  for (int off = 16; off > 0; off >>= 1) ss += __shfl_xor(ss, off, 32);
  float inv = __builtin_amdgcn_rcpf(fmaxf(sqrtf(ss), 1e-12f));
#pragma unroll
  for (int j = 0; j < 4; ++j) outp[n * 128 + tid * 4 + j] = vals[j] * inv;
}

// =======================================================
extern "C" void kernel_launch(void* const* d_in, const int* in_sizes, int n_in,
                              void* d_out, int out_size, void* d_ws, size_t ws_size,
                              hipStream_t stream) {
  const float* x = (const float*)d_in[0];
  float* y1 = (float*)d_ws;                    // [512,6,32,32]
  float* y2 = y1 + 512 * 6 * 32 * 32;          // [512,12,8,8]
  float* y3 = y2 + 512 * 12 * 8 * 8;           // [512,24,4]

  // conv blocks: inputs 1..18 in setup_inputs() insertion order (w,b,g,be,m,v per block)
  conv1_kernel<<<dim3(32, 512), 256, 0, stream>>>(
      x, (const float*)d_in[1], (const float*)d_in[2], (const float*)d_in[3],
      (const float*)d_in[4], (const float*)d_in[5], (const float*)d_in[6], y1);
  conv2_kernel<<<512, 256, 0, stream>>>(
      y1, (const float*)d_in[7], (const float*)d_in[8], (const float*)d_in[9],
      (const float*)d_in[10], (const float*)d_in[11], (const float*)d_in[12], y2);
  conv3_kernel<<<512, 256, 0, stream>>>(
      y2, (const float*)d_in[13], (const float*)d_in[14], (const float*)d_in[15],
      (const float*)d_in[16], (const float*)d_in[17], (const float*)d_in[18], y3);

  // ssm blocks: per block 8 ssm params then cw,cb,g,be,m,v (14 inputs each), starting at 19
  SsmAll P;
  const int Ts[3] = {24, 12, 6};
  const int couts[3] = {12, 6, 8};
  int idx = 19;
  for (int j = 0; j < 3; ++j) {
    SsmParams& q = P.blk[j];
    q.W_in    = (const float*)d_in[idx++];
    q.conv_w  = (const float*)d_in[idx++];
    q.conv_b  = (const float*)d_in[idx++];
    q.dt_bias = (const float*)d_in[idx++];
    q.A_log   = (const float*)d_in[idx++];
    q.D       = (const float*)d_in[idx++];
    q.norm_w  = (const float*)d_in[idx++];
    q.W_out   = (const float*)d_in[idx++];
    q.cw      = (const float*)d_in[idx++];
    q.cb      = (const float*)d_in[idx++];
    q.g       = (const float*)d_in[idx++];
    q.be      = (const float*)d_in[idx++];
    q.m       = (const float*)d_in[idx++];
    q.v       = (const float*)d_in[idx++];
    q.T = Ts[j]; q.cout = couts[j];
  }
  ssm_kernel<<<512, 32, 0, stream>>>(y3, (float*)d_out, P);
}
